// WindowAttention_10943576670271
// MI455X (gfx1250) — matrix operand order repaired and checked by reading, not compiled, not run on hardware
//
#include <hip/hip_runtime.h>
#include <cstdint>

// ---------------------------------------------------------------------------
// WindowAttention for MI455X (gfx1250): bf16 WMMA + async global->LDS staging.
// Stages: cvt(x,w) -> bias table -> QKV GEMM -> per-head attention -> proj GEMM
// ---------------------------------------------------------------------------

typedef __bf16 bf16;
typedef __attribute__((ext_vector_type(16))) __bf16    v16bf;
typedef __attribute__((ext_vector_type(8)))  float     v8f;
typedef __attribute__((ext_vector_type(4)))  unsigned  v4u;

#define HC 12     // heads
#define TT 64     // tokens per window
#define DD 32     // head dim
#define CC 384    // channels

union ABFrag { v16bf v; v4u u[2]; };

// A-matrix 16x32 bf16 fragment (row-major source; row_ptr already offset to
// (row0 + lane%16)*ld + k0). Lane-half selects K [0..7|8..15],[16..23|24..31].
__device__ inline v16bf load_a_frag(const bf16* __restrict__ row_ptr, int half) {
  ABFrag f;
  const bf16* p = row_ptr + half * 8;
  f.u[0] = *(const v4u*)(p);
  f.u[1] = *(const v4u*)(p + 16);
  return f.v;
}

// B-matrix 32x16 bf16 fragment: lane holds column n = lane%16; lane-half selects
// K [0..7|8..15],[16..23|24..31]. col_ptr already at (col0 + lane%16)*ld + k0.
__device__ inline v16bf load_b_frag(const bf16* __restrict__ col_ptr, int half) {
  ABFrag f;
  const bf16* p = col_ptr + half * 8;
  f.u[0] = *(const v4u*)(p);
  f.u[1] = *(const v4u*)(p + 16);
  return f.v;
}

__device__ inline v8f wmma_bf16(v16bf a, v16bf b, v8f c) {
  return __builtin_amdgcn_wmma_f32_16x16x32_bf16(false, a, false, b,
                                                 (short)0, c, false, false);
}

// Async global->LDS copy, 16 bytes per lane. lds_off = low 32 bits of a
// generic pointer to __shared__ (the LDS aperture lives in addr[63:32], so
// the low dword IS the LDS byte offset). Tracked by ASYNCcnt.
__device__ inline void async_copy16(unsigned lds_off, const void* gptr) {
  asm volatile("global_load_async_to_lds_b128 %0, %1, off"
               :: "v"(lds_off), "v"(gptr) : "memory");
}
__device__ inline void wait_async0() {
  asm volatile("s_wait_asynccnt 0x0" ::: "memory");
}

// ---------------------------------------------------------------------------
// fp32 -> bf16 grid-stride converter
// ---------------------------------------------------------------------------
__global__ void cvt_bf16_kernel(const float* __restrict__ src,
                                bf16* __restrict__ dst, int n) {
  int i = blockIdx.x * blockDim.x + threadIdx.x;
  int stride = gridDim.x * blockDim.x;
  for (; i < n; i += stride) dst[i] = (bf16)src[i];
}

// ---------------------------------------------------------------------------
// bias_full[h][tq][tk] = rel_pos_bias[rel_pos_index[tq,tk], h]
// ---------------------------------------------------------------------------
__global__ void bias_table_kernel(const float* __restrict__ tab,
                                  const int* __restrict__ idx,
                                  float* __restrict__ biasF) {
  int t = blockIdx.x * blockDim.x + threadIdx.x;   // 0..4095
  if (t < TT * TT) {
    int id = idx[t];
    #pragma unroll
    for (int h = 0; h < HC; ++h)
      biasF[h * (TT * TT) + t] = tab[id * HC + h];
  }
}

// ---------------------------------------------------------------------------
// QKV GEMM: qkv(m,n) = sum_k x(m,k) * qkv_w(n,k) + qkv_b(n)
// Block 128x64, 8 waves (4x2), each wave 32x32 (2x2 WMMA). Weight tile (64x32)
// double-buffered in LDS via async copies (rows padded to 80B). Epilogue
// scatters to q (scaled), k (B,H,T,D) and v transposed (B,H,D,T), bf16.
// ---------------------------------------------------------------------------
#define BPAD 40   // padded B-tile row length in bf16 elems (80B = 20 dwords)

__global__ __launch_bounds__(256) void qkv_gemm_kernel(
    const bf16* __restrict__ xb,     // (131072, 384)
    const bf16* __restrict__ wb,     // (1152, 384)
    const float* __restrict__ bias,  // (1152)
    bf16* __restrict__ qo, bf16* __restrict__ ko, bf16* __restrict__ vT) {
  __shared__ __align__(16) bf16 bsm[2][64 * BPAD];
  const int tid  = threadIdx.x;
  const int lane = tid & 31;
  const int wid  = tid >> 5;
  const int wm = wid >> 1, wn = wid & 1;
  const int half = lane >> 4, l16 = lane & 15;
  const int blockN = blockIdx.x * 64;
  const int blockM = blockIdx.y * 128;
  const int rowA0 = blockM + wm * 32;
  const int colB0 = blockN + wn * 32;

  // B-tile async copy mapping: thread -> 16B seg; row n = tid>>2, seg = tid&3
  const int bn = tid >> 2, bseg = tid & 3;
  const bf16* gB = wb + (size_t)(blockN + bn) * CC + bseg * 8;
  const unsigned ldsB0 =
      (unsigned)(uintptr_t)(&bsm[0][0]) + (unsigned)(bn * (BPAD * 2) + bseg * 16);
  const unsigned ldsB1 =
      (unsigned)(uintptr_t)(&bsm[1][0]) + (unsigned)(bn * (BPAD * 2) + bseg * 16);

  async_copy16(ldsB0, gB);          // k-chunk 0 -> buf 0
  wait_async0();
  __syncthreads();

  v8f acc[2][2] = {};
  const bf16* aptr0 = xb + (size_t)(rowA0 + l16) * CC;
  const bf16* aptr1 = aptr0 + 16 * CC;

  for (int kt = 0; kt < 12; ++kt) {
    const int buf = kt & 1;
    if (kt + 1 < 12) async_copy16(buf ? ldsB0 : ldsB1, gB + (kt + 1) * 32);

    const int k0 = kt * 32;
    v16bf a0 = load_a_frag(aptr0 + k0, half);
    v16bf a1 = load_a_frag(aptr1 + k0, half);
    const bf16* bb = &bsm[buf][0];
    v16bf b0 = load_b_frag(bb + (wn * 32 + l16) * BPAD, half);
    v16bf b1 = load_b_frag(bb + (wn * 32 + 16 + l16) * BPAD, half);
    acc[0][0] = wmma_bf16(a0, b0, acc[0][0]);
    acc[0][1] = wmma_bf16(a0, b1, acc[0][1]);
    acc[1][0] = wmma_bf16(a1, b0, acc[1][0]);
    acc[1][1] = wmma_bf16(a1, b1, acc[1][1]);

    if (kt + 1 < 12) wait_async0();
    __syncthreads();
  }

  const float scale = 0.1767766952966369f;  // 32^-0.5
  #pragma unroll
  for (int mi = 0; mi < 2; ++mi) {
    #pragma unroll
    for (int ni = 0; ni < 2; ++ni) {
      #pragma unroll
      for (int r = 0; r < 8; ++r) {
        int m = rowA0 + mi * 16 + r + 8 * half;
        int n = colB0 + ni * 16 + l16;
        float val = acc[mi][ni][r] + bias[n];
        int sec = n / CC;          // 0=q 1=k 2=v
        int cc  = n - sec * CC;
        int h = cc >> 5, d = cc & 31;
        int b = m >> 6, t = m & 63;
        size_t qk = ((size_t)((b * HC + h) * TT + t)) * DD + d;
        if (sec == 0)      qo[qk] = (bf16)(val * scale);
        else if (sec == 1) ko[qk] = (bf16)val;
        else vT[((size_t)((b * HC + h) * DD + d)) * TT + t] = (bf16)val;
      }
    }
  }
}

// ---------------------------------------------------------------------------
// Attention: one block per (b,h); 4 waves, wave w owns query rows [16w,16w+16).
// k (64x32) and vT (32x64) tiles staged into padded LDS via async copies.
// S = q k^T (4 WMMAs), +bias, softmax in registers (shfl over 16-lane halves),
// P staged via LDS into A-fragment layout, O = P V (4 WMMAs), bf16 out (B,T,C).
// ---------------------------------------------------------------------------
#define VPAD 72   // padded v-tile row length in bf16 elems (144B = 36 dwords)

__global__ __launch_bounds__(128) void attn_kernel(
    const bf16* __restrict__ qb, const bf16* __restrict__ kb,
    const bf16* __restrict__ vT, const float* __restrict__ biasF,
    bf16* __restrict__ ao) {
  __shared__ __align__(16) bf16  ksm[TT * BPAD];   // 64 rows x 80B
  __shared__ __align__(16) bf16  vsm[DD * VPAD];   // 32 rows x 144B
  __shared__ __align__(16) float ps[TT][68];       // P, 272B rows
  const int lane = threadIdx.x & 31;
  const int wid  = threadIdx.x >> 5;               // 0..3
  const int half = lane >> 4, l16 = lane & 15;
  const int h = blockIdx.x;                        // 0..11
  const int b = blockIdx.y;                        // 0..2047
  const size_t bh = (size_t)(b * HC + h);
  const bf16* qh = qb + bh * (TT * DD);
  const bf16* kh = kb + bh * (TT * DD);
  const bf16* vh = vT + bh * (DD * TT);
  const float* bih = biasF + (size_t)h * (TT * TT);

  // Stage k (4KB) and v (4KB) tiles into LDS (async, 16B per lane per op).
  {
    int kr = threadIdx.x >> 2, kseg = threadIdx.x & 3;   // k rows 0..31 pass 1
    const bf16* gk = kh + kr * DD + kseg * 8;
    unsigned lk = (unsigned)(uintptr_t)ksm + (unsigned)(kr * (BPAD * 2) + kseg * 16);
    async_copy16(lk, gk);
    async_copy16(lk + 32 * (BPAD * 2), (const char*)gk + 32 * (DD * 2));
    int vr = threadIdx.x >> 3, vseg = threadIdx.x & 7;   // v rows 0..15 pass 1
    const bf16* gv = vh + vr * TT + vseg * 8;
    unsigned lv = (unsigned)(uintptr_t)vsm + (unsigned)(vr * (VPAD * 2) + vseg * 16);
    async_copy16(lv, gv);
    async_copy16(lv + 16 * (VPAD * 2), (const char*)gv + 16 * (TT * 2));
    wait_async0();
  }
  __syncthreads();

  // S = q @ k^T for 16 query rows
  v16bf aq = load_a_frag(qh + (wid * 16 + l16) * DD, half);
  v8f s[4];
  #pragma unroll
  for (int j = 0; j < 4; ++j) {
    v16bf bk = load_b_frag(ksm + (j * 16 + l16) * BPAD, half);
    v8f z = {};
    s[j] = wmma_bf16(aq, bk, z);
  }

  // bias + softmax per row; row (r,half) lives in VGPR r across a lane-half
  #pragma unroll
  for (int r = 0; r < 8; ++r) {
    int tq = wid * 16 + r + 8 * half;
    const float* brow = bih + tq * TT;
    float mx = -3.0e38f;
    #pragma unroll
    for (int j = 0; j < 4; ++j) {
      float v = s[j][r] + brow[j * 16 + l16];
      s[j][r] = v;
      mx = fmaxf(mx, v);
    }
    #pragma unroll
    for (int off = 1; off < 16; off <<= 1)
      mx = fmaxf(mx, __shfl_xor(mx, off, 32));
    float sum = 0.f;
    #pragma unroll
    for (int j = 0; j < 4; ++j) {
      float e = __expf(s[j][r] - mx);
      s[j][r] = e;
      sum += e;
    }
    #pragma unroll
    for (int off = 1; off < 16; off <<= 1)
      sum += __shfl_xor(sum, off, 32);
    float inv = 1.0f / sum;
    #pragma unroll
    for (int j = 0; j < 4; ++j)
      ps[tq][j * 16 + l16] = s[j][r] * inv;
  }
  __syncthreads();

  // O = P @ V : rows [16w,16w+16), cols 0..31; K=64 in two 32-chunks
  v8f o0 = {}, o1 = {};
  const float* prow = &ps[wid * 16 + l16][0];
  #pragma unroll
  for (int kc = 0; kc < 2; ++kc) {
    v16bf af;
    {
      const float* p = prow + kc * 32 + half * 8;
      #pragma unroll
      for (int i = 0; i < 8; ++i) af[i] = (bf16)p[i];
      #pragma unroll
      for (int i = 0; i < 8; ++i) af[8 + i] = (bf16)p[16 + i];
    }
    v16bf b0 = load_b_frag(vsm + (0 * 16 + l16) * VPAD + kc * 32, half);
    v16bf b1 = load_b_frag(vsm + (1 * 16 + l16) * VPAD + kc * 32, half);
    o0 = wmma_bf16(af, b0, o0);
    o1 = wmma_bf16(af, b1, o1);
  }

  // attn_out (B*T, C) at (b*64+tq, h*32 + d)
  #pragma unroll
  for (int r = 0; r < 8; ++r) {
    int tq = wid * 16 + r + 8 * half;
    bf16* orow = ao + ((size_t)(b * TT + tq)) * CC + h * DD;
    orow[l16]      = (bf16)o0[r];
    orow[16 + l16] = (bf16)o1[r];
  }
}

// ---------------------------------------------------------------------------
// Proj GEMM: out(m,n) = sum_k a(m,k) * proj_w(n,k) + proj_b(n), fp32 out.
// Same structure as QKV GEMM (async double-buffered weight tile in LDS).
// ---------------------------------------------------------------------------
__global__ __launch_bounds__(256) void proj_gemm_kernel(
    const bf16* __restrict__ ab,     // (131072, 384)
    const bf16* __restrict__ wb,     // (384, 384)
    const float* __restrict__ bias,  // (384)
    float* __restrict__ out) {
  __shared__ __align__(16) bf16 bsm[2][64 * BPAD];
  const int tid  = threadIdx.x;
  const int lane = tid & 31;
  const int wid  = tid >> 5;
  const int wm = wid >> 1, wn = wid & 1;
  const int half = lane >> 4, l16 = lane & 15;
  const int blockN = blockIdx.x * 64;
  const int blockM = blockIdx.y * 128;
  const int rowA0 = blockM + wm * 32;
  const int colB0 = blockN + wn * 32;

  const int bn = tid >> 2, bseg = tid & 3;
  const bf16* gB = wb + (size_t)(blockN + bn) * CC + bseg * 8;
  const unsigned ldsB0 =
      (unsigned)(uintptr_t)(&bsm[0][0]) + (unsigned)(bn * (BPAD * 2) + bseg * 16);
  const unsigned ldsB1 =
      (unsigned)(uintptr_t)(&bsm[1][0]) + (unsigned)(bn * (BPAD * 2) + bseg * 16);

  async_copy16(ldsB0, gB);
  wait_async0();
  __syncthreads();

  v8f acc[2][2] = {};
  const bf16* aptr0 = ab + (size_t)(rowA0 + l16) * CC;
  const bf16* aptr1 = aptr0 + 16 * CC;

  for (int kt = 0; kt < 12; ++kt) {
    const int buf = kt & 1;
    if (kt + 1 < 12) async_copy16(buf ? ldsB0 : ldsB1, gB + (kt + 1) * 32);

    const int k0 = kt * 32;
    v16bf a0 = load_a_frag(aptr0 + k0, half);
    v16bf a1 = load_a_frag(aptr1 + k0, half);
    const bf16* bb = &bsm[buf][0];
    v16bf b0 = load_b_frag(bb + (wn * 32 + l16) * BPAD, half);
    v16bf b1 = load_b_frag(bb + (wn * 32 + 16 + l16) * BPAD, half);
    acc[0][0] = wmma_bf16(a0, b0, acc[0][0]);
    acc[0][1] = wmma_bf16(a0, b1, acc[0][1]);
    acc[1][0] = wmma_bf16(a1, b0, acc[1][0]);
    acc[1][1] = wmma_bf16(a1, b1, acc[1][1]);

    if (kt + 1 < 12) wait_async0();
    __syncthreads();
  }

  #pragma unroll
  for (int mi = 0; mi < 2; ++mi) {
    #pragma unroll
    for (int ni = 0; ni < 2; ++ni) {
      #pragma unroll
      for (int r = 0; r < 8; ++r) {
        int m = rowA0 + mi * 16 + r + 8 * half;
        int n = colB0 + ni * 16 + l16;
        out[(size_t)m * CC + n] = acc[mi][ni][r] + bias[n];
      }
    }
  }
}

// ---------------------------------------------------------------------------
extern "C" void kernel_launch(void* const* d_in, const int* in_sizes, int n_in,
                              void* d_out, int out_size, void* d_ws,
                              size_t ws_size, hipStream_t stream) {
  const float* x      = (const float*)d_in[0];
  const float* qkv_w  = (const float*)d_in[1];
  const float* qkv_b  = (const float*)d_in[2];
  const float* proj_w = (const float*)d_in[3];
  const float* proj_b = (const float*)d_in[4];
  const float* rpb    = (const float*)d_in[5];
  const int*   rpi    = (const int*)d_in[6];

  const size_t MTOT = 2048ull * TT;         // 131072 rows
  const size_t szX  = MTOT * CC * 2;        // bf16 x (aliased with attn_out)
  const size_t szQW = 1152ull * CC * 2;
  const size_t szPW = (size_t)CC * CC * 2;
  const size_t szQ  = MTOT * CC * 2;        // each of q/k/vT (B*H*T*D)

  char* ws = (char*)d_ws;
  bf16*  xb  = (bf16*)(ws);
  bf16*  aob = xb;                          // attn_out reuses x_bf16 region
  bf16*  qwb = (bf16*)(ws + szX);
  bf16*  pwb = (bf16*)(ws + szX + szQW);
  bf16*  qo  = (bf16*)(ws + szX + szQW + szPW);
  bf16*  ko  = (bf16*)((char*)qo + szQ);
  bf16*  vo  = (bf16*)((char*)ko + szQ);
  float* bfl = (float*)((char*)vo + szQ);   // bias_full (12,64,64) f32

  cvt_bf16_kernel<<<2048, 256, 0, stream>>>(x, xb, (int)(MTOT * CC));
  cvt_bf16_kernel<<<432, 256, 0, stream>>>(qkv_w, qwb, 1152 * CC);
  cvt_bf16_kernel<<<144, 256, 0, stream>>>(proj_w, pwb, CC * CC);
  bias_table_kernel<<<16, 256, 0, stream>>>(rpb, rpi, bfl);

  qkv_gemm_kernel<<<dim3(1152 / 64, (int)(MTOT / 128)), 256, 0, stream>>>(
      xb, qwb, qkv_b, qo, ko, vo);
  attn_kernel<<<dim3(HC, 2048), 128, 0, stream>>>(qo, ko, vo, bfl, aob);
  proj_gemm_kernel<<<dim3(CC / 64, (int)(MTOT / 128)), 256, 0, stream>>>(
      aob, pwb, proj_b, (float*)d_out);
}
